// gLSTMCell_11252814315783
// MI455X (gfx1250) — matrix operand order repaired and checked by reading, not compiled, not run on hardware
//
#include <hip/hip_runtime.h>
#include <hip/hip_bf16.h>

typedef __attribute__((ext_vector_type(8)))  _Float16 v8h;
typedef __attribute__((ext_vector_type(16))) _Float16 v16h;
typedef __attribute__((ext_vector_type(8)))  float    v8f;

#define NB 32      // batch N
#define VV 5000    // vertices
#define GG 16      // groups
#define II 64      // input dim
#define OO 64      // output dim
#define KK 128     // combined K = I + O

// ---------------------------------------------------------------------------
// Prologue: pack weight_ih|weight_hh -> f16 table W[gate][group][o][k], k-major
// so each WMMA B-fragment lane reads two contiguous 16B chunks.
// ---------------------------------------------------------------------------
__global__ void pack_weights_f16(const float* __restrict__ w_ih,
                                 const float* __restrict__ w_hh,
                                 _Float16* __restrict__ wpk) {
  int idx = blockIdx.x * blockDim.x + threadIdx.x;   // over 4*16*64*128
  if (idx >= 4 * GG * OO * KK) return;
  int k    = idx & (KK - 1);
  int rest = idx >> 7;            // (gate*GG + group)*OO + o
  int o    = rest & (OO - 1);
  int ggr  = rest >> 6;           // gate*GG + group
  float val = (k < II) ? w_ih[((size_t)ggr * OO + o) * II + k]
                       : w_hh[((size_t)ggr * OO + o) * OO + (k - II)];
  wpk[idx] = (_Float16)val;
}

// ---------------------------------------------------------------------------
// Main: one 256-thread workgroup (8 wave32) per vertex v.
//   waves: gate = wv/2 (0..3), M-tile = wv%2 (rows 0-15 / 16-31)
//   each wave: 4 N-tiles x 4 K-steps of v_wmma_f32_16x16x32_f16
// ---------------------------------------------------------------------------
__global__ __launch_bounds__(256)
void glstm_cell_wmma(const float* __restrict__ x,
                     const float* __restrict__ h,
                     const float* __restrict__ c,
                     const int*   __restrict__ gidx,
                     const float* __restrict__ b_ih,
                     const float* __restrict__ b_hh,
                     const _Float16* __restrict__ wpk,
                     float* __restrict__ h_out,
                     float* __restrict__ c_out) {
  __shared__ _Float16 ldsA[NB][KK];       // 8 KB : [x | h] as f16
  __shared__ float    ldsG[4][NB][OO];    // 32 KB: gate pre-activations

  const int v   = blockIdx.x;
  const int tid = threadIdx.x;
  const int gr  = gidx[v];                // uniform -> scalar load

  // ---- stage A-panel (32 x 128) into LDS as f16 ----------------------------
  for (int e = tid; e < NB * KK; e += 256) {
    int n = e >> 7;
    int k = e & (KK - 1);
    float val = (k < II)
        ? x[(size_t)n * VV * II + (size_t)v * II + k]
        : h[(size_t)n * VV * OO + (size_t)v * OO + (k - II)];
    ldsA[n][k] = (_Float16)val;
  }
  __syncthreads();

  // ---- WMMA stage ----------------------------------------------------------
  const int  wv     = tid >> 5;
  const int  lane   = tid & 31;
  const int  gate   = wv >> 1;
  const int  m0     = (wv & 1) * 16;
  const int  ln     = lane & 15;
  const bool hiHalf = lane >= 16;
  const int  m      = m0 + ln;

  v8f acc0 = {}, acc1 = {}, acc2 = {}, acc3 = {};

  const size_t wbase = (((size_t)gate * GG + gr) * OO) * KK;

  #pragma unroll
  for (int kk = 0; kk < 4; ++kk) {
    // A fragment (16x32 f16): lane<16 holds K {0..7, 16..23} of the step,
    // lane>=16 holds K {8..15, 24..31}.
    const int akb = kk * 32 + (hiHalf ? 8 : 0);
    const _Float16* ap = &ldsA[m][akb];
    v8h alo = *(const v8h*)(ap);
    v8h ahi = *(const v8h*)(ap + 16);
    v16h afrag = __builtin_shufflevector(alo, ahi,
        0,1,2,3,4,5,6,7,8,9,10,11,12,13,14,15);

    // B fragment (32x16 f16): lane<16 = K {0..7, 16..23}, lane>=16 = K {8..15, 24..31} of its column.
    const int bkb = kk * 32 + (hiHalf ? 8 : 0);

    #pragma unroll
    for (int nt = 0; nt < 4; ++nt) {
      const int o = nt * 16 + ln;
      const _Float16* bp = &wpk[wbase + (size_t)o * KK + bkb];
      v8h blo = *(const v8h*)(bp);
      v8h bhi = *(const v8h*)(bp + 16);
      v16h bfrag = __builtin_shufflevector(blo, bhi,
          0,1,2,3,4,5,6,7,8,9,10,11,12,13,14,15);

      switch (nt) {
        case 0: acc0 = __builtin_amdgcn_wmma_f32_16x16x32_f16(
                    false, afrag, false, bfrag, (short)0, acc0, false, false); break;
        case 1: acc1 = __builtin_amdgcn_wmma_f32_16x16x32_f16(
                    false, afrag, false, bfrag, (short)0, acc1, false, false); break;
        case 2: acc2 = __builtin_amdgcn_wmma_f32_16x16x32_f16(
                    false, afrag, false, bfrag, (short)0, acc2, false, false); break;
        default: acc3 = __builtin_amdgcn_wmma_f32_16x16x32_f16(
                    false, afrag, false, bfrag, (short)0, acc3, false, false); break;
      }
    }
  }

  // ---- park gate results in LDS (C layout: VGPR r, lane L -> M=r+(L<16?0:8), N=L%16)
  const int mrow = m0 + (hiHalf ? 8 : 0);
  #pragma unroll
  for (int r = 0; r < 8; ++r) {
    ldsG[gate][mrow + r][ 0 + ln] = acc0[r];
    ldsG[gate][mrow + r][16 + ln] = acc1[r];
    ldsG[gate][mrow + r][32 + ln] = acc2[r];
    ldsG[gate][mrow + r][48 + ln] = acc3[r];
  }
  __syncthreads();

  // ---- elementwise LSTM combine -------------------------------------------
  for (int e = tid; e < NB * OO; e += 256) {
    int n = e >> 6;
    int o = e & (OO - 1);

    float z[4];
    #pragma unroll
    for (int g = 0; g < 4; ++g) {
      int bi = (g * GG + gr) * OO + o;
      z[g] = ldsG[g][n][o] + b_ih[bi] + b_hh[bi];
    }
    float ig = 1.0f / (1.0f + expf(-z[0]));
    float fg = 1.0f / (1.0f + expf(-z[1]));
    float gg = tanhf(z[2]);
    float og = 1.0f / (1.0f + expf(-z[3]));

    size_t idx = (size_t)n * VV * OO + (size_t)v * OO + o;
    float cn = fg * c[idx] + ig * gg;
    float hn = og * tanhf(cn);
    h_out[idx] = hn;
    c_out[idx] = cn;
  }
}

// ---------------------------------------------------------------------------
extern "C" void kernel_launch(void* const* d_in, const int* in_sizes, int n_in,
                              void* d_out, int out_size, void* d_ws, size_t ws_size,
                              hipStream_t stream) {
  const float* x     = (const float*)d_in[0];
  const float* h     = (const float*)d_in[1];
  const float* c     = (const float*)d_in[2];
  const int*   gi    = (const int*)  d_in[3];
  const float* w_ih  = (const float*)d_in[4];
  const float* b_ih  = (const float*)d_in[5];
  const float* w_hh  = (const float*)d_in[6];
  const float* b_hh  = (const float*)d_in[7];

  _Float16* wpk = (_Float16*)d_ws;          // 4*16*64*128 f16 = 1 MB
  float* h_out = (float*)d_out;
  float* c_out = h_out + (size_t)NB * VV * OO;

  const int wtot = 4 * GG * OO * KK;
  pack_weights_f16<<<(wtot + 255) / 256, 256, 0, stream>>>(w_ih, w_hh, wpk);
  glstm_cell_wmma<<<VV, 256, 0, stream>>>(x, h, c, gi, b_ih, b_hh, wpk,
                                          h_out, c_out);
}
